// MultiHeadAttention_77601469104947
// MI455X (gfx1250) — matrix operand order repaired and checked by reading, not compiled, not run on hardware
//
#include <hip/hip_runtime.h>
#include <hip/hip_bf16.h>

// ---------------------------------------------------------------------------
// MHA forward for MI455X (gfx1250): all GEMM stages on v_wmma_f32_16x16x32_f16
// Fixed sizes: B=4, S=2048, D_MODEL=1024, H=16, HD=64
// ---------------------------------------------------------------------------

typedef _Float16 half_t;
typedef __attribute__((ext_vector_type(16))) _Float16 v16h;
typedef __attribute__((ext_vector_type(8)))  _Float16 v8h;
typedef __attribute__((ext_vector_type(8)))  float    v8f;

union Frag16 { v16h v; v8h h[2]; };

// A-matrix fragment (16 rows x 32 K, f16). Per ISA: lanes 0-15 hold M=0..15
// with K runs {k0..k0+7, k0+16..k0+23}; lanes 16-31 hold M=0..15 with
// K runs {k0+8..15, k0+24..31}.  ld in halves.
__device__ inline v16h load_frag_a(const half_t* src, int row0, int k0, int ld) {
  const int lane = threadIdx.x & 31;
  const int idx  = lane & 15;
  const int hf   = lane >> 4;
  const half_t* p = src + (size_t)(row0 + idx) * ld + k0 + hf * 8;
  Frag16 f;
  f.h[0] = *(const v8h*)(p);
  f.h[1] = *(const v8h*)(p + 16);
  return f.v;
}

// B-matrix fragment (32 K x 16 cols, f16). Same K order as A: lanes 0-15 =
// column idx, K runs {k0..k0+7, k0+16..k0+23}; lanes 16-31 = column idx,
// K runs {k0+8..15, k0+24..31}.  Source is row-major with row n == column n of B.
__device__ inline v16h load_frag_b(const half_t* src, int col0, int k0, int ld) {
  const int lane = threadIdx.x & 31;
  const int idx  = lane & 15;
  const int hf   = lane >> 4;
  const half_t* p = src + (size_t)(col0 + idx) * ld + k0 + hf * 8;
  Frag16 f;
  f.h[0] = *(const v8h*)(p);
  f.h[1] = *(const v8h*)(p + 16);
  return f.v;
}

__device__ inline v8f wmma_f16(v16h a, v16h b, v8f c) {
  return __builtin_amdgcn_wmma_f32_16x16x32_f16(
      /*neg_a=*/false, a, /*neg_b=*/false, b,
      /*c_mod=*/(short)0, c, /*reuse_a=*/false, /*reuse_b=*/false);
}

// ---------------------------------------------------------------------------
// fp32 -> f16 cast (grid-stride)
// ---------------------------------------------------------------------------
__global__ void cast_f32_f16(const float* __restrict__ src,
                             half_t* __restrict__ dst, int n) {
  for (int i = blockIdx.x * blockDim.x + threadIdx.x; i < n;
       i += gridDim.x * blockDim.x)
    dst[i] = (half_t)src[i];
}

// ---------------------------------------------------------------------------
// Mask prep: maskf[b][k] = masked ? -1e30 : 0, with the "all masked -> unmask
// key 0" fix.  One block per batch.
// ---------------------------------------------------------------------------
__global__ void mask_prep(const unsigned char* __restrict__ mask,
                          float* __restrict__ maskf) {
  const int K = 2048;
  const int b = blockIdx.x;
  const int tid = threadIdx.x;
  __shared__ int sAll[256];
  int allm = 1;
  for (int j = tid; j < K; j += 256) allm &= (mask[b * K + j] != 0) ? 1 : 0;
  sAll[tid] = allm;
  __syncthreads();
  for (int s = 128; s > 0; s >>= 1) {
    if (tid < s) sAll[tid] &= sAll[tid + s];
    __syncthreads();
  }
  const int all = sAll[0];
  for (int j = tid; j < K; j += 256) {
    bool m = mask[b * K + j] != 0;
    if (j == 0 && all) m = false;
    maskf[b * K + j] = m ? -1e30f : 0.0f;
  }
}

// ---------------------------------------------------------------------------
// GEMM: C[M,N] = A[M,K](f16) x W[N,K]^T(f16) + bias[N], fp32 out.
// Block = 256 threads = 8 waves; block tile 128x128; wave tile 32x64.
// grid = (N/128, M/128)
// ---------------------------------------------------------------------------
__global__ __launch_bounds__(256) void gemm_bias_wmma(
    const half_t* __restrict__ A, const half_t* __restrict__ W,
    const float* __restrict__ bias, float* __restrict__ C,
    int M, int N, int K) {
  const int tid  = threadIdx.x;
  const int w    = tid >> 5;
  const int lane = tid & 31;
  const int idx  = lane & 15;
  const int hf   = lane >> 4;
  const int wm   = w & 3;   // 4 waves along M
  const int wn   = w >> 2;  // 2 waves along N
  const int row0 = blockIdx.y * 128 + wm * 32;
  const int col0 = blockIdx.x * 128 + wn * 64;

  v8f acc[2][4] = {};
  for (int k0 = 0; k0 < K; k0 += 32) {
    v16h a0 = load_frag_a(A, row0,      k0, K);
    v16h a1 = load_frag_a(A, row0 + 16, k0, K);
#pragma unroll
    for (int j = 0; j < 4; ++j) {
      v16h bf = load_frag_b(W, col0 + j * 16, k0, K);
      acc[0][j] = wmma_f16(a0, bf, acc[0][j]);
      acc[1][j] = wmma_f16(a1, bf, acc[1][j]);
    }
  }
#pragma unroll
  for (int i = 0; i < 2; ++i)
#pragma unroll
    for (int j = 0; j < 4; ++j) {
      const int col = col0 + j * 16 + idx;
      const float bv = bias[col];
#pragma unroll
      for (int r = 0; r < 8; ++r) {
        const int row = row0 + i * 16 + r + hf * 8;
        C[(size_t)row * N + col] = acc[i][j][r] + bv;
      }
    }
}

// ---------------------------------------------------------------------------
// RoPE + repack: proj fp32 [B,S,1024] -> dst f16 [B,H,S,64] with rotation.
// One thread per (b,h,s,pair); 4*16*2048*32 threads.
// ---------------------------------------------------------------------------
__global__ void rope_pack(const float* __restrict__ proj,
                          half_t* __restrict__ dst) {
  const int i  = blockIdx.x * blockDim.x + threadIdx.x;
  const int dp = i & 31;
  const int s  = (i >> 5) & 2047;
  const int h  = (i >> 16) & 15;
  const int b  = i >> 20;
  // inv_freq = 10000^(-(2*dp)/64)
  const float inv = __expf(-(float)(2 * dp) * (9.210340371976184f / 64.0f));
  const float ang = (float)s * inv;
  float sn, cs;
  __sincosf(ang, &sn, &cs);
  const float* p = proj + ((size_t)(b * 2048 + s)) * 1024 + h * 64 + 2 * dp;
  const float x1 = p[0], x2 = p[1];
  half_t* q = dst + ((size_t)((b * 16 + h) * 2048 + s)) * 64 + 2 * dp;
  q[0] = (half_t)(x1 * cs - x2 * sn);
  q[1] = (half_t)(x1 * sn + x2 * cs);
}

// ---------------------------------------------------------------------------
// V repack with transpose: proj fp32 [B,S,1024] -> vh f16 [B,H,64,S]
// (d-major so attention V B-fragments are contiguous over keys).
// One thread per element; 4*16*64*2048 threads.
// ---------------------------------------------------------------------------
__global__ void vpack_t(const float* __restrict__ proj,
                        half_t* __restrict__ vh) {
  const int i = blockIdx.x * blockDim.x + threadIdx.x;
  const int s = i & 2047;
  const int d = (i >> 11) & 63;
  const int h = (i >> 17) & 15;
  const int b = i >> 21;
  const float x = proj[((size_t)(b * 2048 + s)) * 1024 + h * 64 + d];
  vh[((size_t)((b * 16 + h) * 64 + d)) * 2048 + s] = (half_t)x;
}

// ---------------------------------------------------------------------------
// Flash attention: grid (S/64, B*H), 128 threads (4 waves).
// Each wave: 16 query rows x 64 head dims. Per 64-key tile:
//   8 wmma (Q.K^T) -> online softmax -> P via per-wave LDS -> 8 wmma (P.V)
// Output: attnh f16 [B,S,1024] (row = q position, col = h*64+d).
// ---------------------------------------------------------------------------
__global__ __launch_bounds__(128) void flash_attn_wmma(
    const half_t* __restrict__ qh, const half_t* __restrict__ kh,
    const half_t* __restrict__ vh, const float* __restrict__ maskf,
    half_t* __restrict__ attnh) {
  __shared__ half_t Pl[4][16][72];  // per-wave P staging, padded pitch
  const int tid  = threadIdx.x;
  const int w    = tid >> 5;
  const int lane = tid & 31;
  const int idx  = lane & 15;
  const int hf   = lane >> 4;
  const int bh   = blockIdx.y;  // b*16 + h
  const int b    = bh >> 4;
  const int h    = bh & 15;
  const int qrow0 = blockIdx.x * 64 + w * 16;

  const half_t* qb = qh + (size_t)bh * 2048 * 64;
  const half_t* kb = kh + (size_t)bh * 2048 * 64;
  const half_t* vb = vh + (size_t)bh * 64 * 2048;
  const float* mrowp = maskf + b * 2048;

  const v16h qf0 = load_frag_a(qb, qrow0, 0, 64);
  const v16h qf1 = load_frag_a(qb, qrow0, 32, 64);

  v8f o[4] = {};
  float mx[8], l[8];
#pragma unroll
  for (int r = 0; r < 8; ++r) { mx[r] = -1e30f; l[r] = 0.0f; }

  for (int kt = 0; kt < 2048; kt += 64) {
    // ---- scores S = (Q K^T) / 8 + mask ----
    v8f s[4] = {};
#pragma unroll
    for (int jt = 0; jt < 4; ++jt) {
      v16h kf0 = load_frag_b(kb, kt + jt * 16, 0, 64);
      v16h kf1 = load_frag_b(kb, kt + jt * 16, 32, 64);
      s[jt] = wmma_f16(qf0, kf0, s[jt]);
      s[jt] = wmma_f16(qf1, kf1, s[jt]);
    }
    float mk[4];
#pragma unroll
    for (int jt = 0; jt < 4; ++jt) mk[jt] = mrowp[kt + jt * 16 + idx];
#pragma unroll
    for (int jt = 0; jt < 4; ++jt)
#pragma unroll
      for (int r = 0; r < 8; ++r)
        s[jt][r] = s[jt][r] * 0.125f + mk[jt];

    // ---- online softmax (row r lives in VGPR r across a 16-lane half) ----
#pragma unroll
    for (int r = 0; r < 8; ++r) {
      float rm = fmaxf(fmaxf(s[0][r], s[1][r]), fmaxf(s[2][r], s[3][r]));
      rm = fmaxf(rm, __shfl_xor(rm, 1, 32));
      rm = fmaxf(rm, __shfl_xor(rm, 2, 32));
      rm = fmaxf(rm, __shfl_xor(rm, 4, 32));
      rm = fmaxf(rm, __shfl_xor(rm, 8, 32));
      const float mn = fmaxf(mx[r], rm);
      const float alpha = __expf(mx[r] - mn);
      mx[r] = mn;
      float rs = 0.0f;
#pragma unroll
      for (int jt = 0; jt < 4; ++jt) {
        const float p = __expf(s[jt][r] - mn);
        s[jt][r] = p;
        rs += p;
      }
      rs += __shfl_xor(rs, 1, 32);
      rs += __shfl_xor(rs, 2, 32);
      rs += __shfl_xor(rs, 4, 32);
      rs += __shfl_xor(rs, 8, 32);
      l[r] = l[r] * alpha + rs;
#pragma unroll
      for (int jt = 0; jt < 4; ++jt) o[jt][r] *= alpha;
    }

    // ---- P: C-layout -> LDS (f16) -> A-layout fragments ----
#pragma unroll
    for (int jt = 0; jt < 4; ++jt)
#pragma unroll
      for (int r = 0; r < 8; ++r)
        Pl[w][r + hf * 8][jt * 16 + idx] = (half_t)s[jt][r];
    const v16h pf0 = load_frag_a(&Pl[w][0][0], 0, 0, 72);
    const v16h pf1 = load_frag_a(&Pl[w][0][0], 0, 32, 72);

    // ---- O += P V ----
#pragma unroll
    for (int jt = 0; jt < 4; ++jt) {
      v16h vf0 = load_frag_b(vb, jt * 16, kt, 2048);
      v16h vf1 = load_frag_b(vb, jt * 16, kt + 32, 2048);
      o[jt] = wmma_f16(pf0, vf0, o[jt]);
      o[jt] = wmma_f16(pf1, vf1, o[jt]);
    }
  }

  // ---- normalize + store f16 [B,S,1024] ----
#pragma unroll
  for (int jt = 0; jt < 4; ++jt)
#pragma unroll
    for (int r = 0; r < 8; ++r) {
      const int row = qrow0 + r + hf * 8;
      attnh[(size_t)(b * 2048 + row) * 1024 + h * 64 + jt * 16 + idx] =
          (half_t)(o[jt][r] / l[r]);
    }
}

// ---------------------------------------------------------------------------
// Launch
// ---------------------------------------------------------------------------
extern "C" void kernel_launch(void* const* d_in, const int* in_sizes, int n_in,
                              void* d_out, int out_size, void* d_ws, size_t ws_size,
                              hipStream_t stream) {
  (void)in_sizes; (void)n_in; (void)out_size; (void)ws_size;
  const float* query = (const float*)d_in[0];
  const float* key   = (const float*)d_in[1];
  const float* value = (const float*)d_in[2];
  const unsigned char* mask = (const unsigned char*)d_in[3];
  const float* Wq = (const float*)d_in[4];
  const float* bq = (const float*)d_in[5];
  const float* Wk = (const float*)d_in[6];
  const float* bk = (const float*)d_in[7];
  const float* Wv = (const float*)d_in[8];
  const float* bv = (const float*)d_in[9];
  const float* Wo = (const float*)d_in[10];
  const float* bo = (const float*)d_in[11];
  float* out = (float*)d_out;

  const size_t DM = 1024, S = 2048, B = 4;
  const size_t ROWS = B * S;  // 8192

  char* ws = (char*)d_ws;
  size_t off = 0;
  auto alloc = [&](size_t bytes) -> void* {
    void* p = ws + off;
    off += (bytes + 255) & ~(size_t)255;
    return p;
  };
  half_t* wqh   = (half_t*)alloc(DM * DM * 2);
  half_t* wkh   = (half_t*)alloc(DM * DM * 2);
  half_t* wvh   = (half_t*)alloc(DM * DM * 2);
  half_t* woh   = (half_t*)alloc(DM * DM * 2);
  half_t* xh    = (half_t*)alloc(ROWS * DM * 2);
  float*  proj  = (float*) alloc(ROWS * DM * 4);
  half_t* qhb   = (half_t*)alloc(ROWS * DM * 2);
  half_t* khb   = (half_t*)alloc(ROWS * DM * 2);
  half_t* vhb   = (half_t*)alloc(ROWS * DM * 2);
  half_t* attnh = (half_t*)alloc(ROWS * DM * 2);
  float*  maskf = (float*) alloc(B * S * 4);

  const int nW = (int)(DM * DM);        // 1,048,576
  const int nX = (int)(ROWS * DM);      // 8,388,608
  cast_f32_f16<<<nW / 256, 256, 0, stream>>>(Wq, wqh, nW);
  cast_f32_f16<<<nW / 256, 256, 0, stream>>>(Wk, wkh, nW);
  cast_f32_f16<<<nW / 256, 256, 0, stream>>>(Wv, wvh, nW);
  cast_f32_f16<<<nW / 256, 256, 0, stream>>>(Wo, woh, nW);
  mask_prep<<<4, 256, 0, stream>>>(mask, maskf);

  const dim3 gemmGrid(8, 64);           // N/128, M/128
  const int ropeBlocks = (4 * 16 * 2048 * 32) / 256;   // 16384
  const int vpackBlocks = nX / 256;                    // 32768

  // Q projection -> RoPE pack
  cast_f32_f16<<<nX / 256, 256, 0, stream>>>(query, xh, nX);
  gemm_bias_wmma<<<gemmGrid, 256, 0, stream>>>(xh, wqh, bq, proj, 8192, 1024, 1024);
  rope_pack<<<ropeBlocks, 256, 0, stream>>>(proj, qhb);

  // K projection -> RoPE pack
  cast_f32_f16<<<nX / 256, 256, 0, stream>>>(key, xh, nX);
  gemm_bias_wmma<<<gemmGrid, 256, 0, stream>>>(xh, wkh, bk, proj, 8192, 1024, 1024);
  rope_pack<<<ropeBlocks, 256, 0, stream>>>(proj, khb);

  // V projection -> transposed pack
  cast_f32_f16<<<nX / 256, 256, 0, stream>>>(value, xh, nX);
  gemm_bias_wmma<<<gemmGrid, 256, 0, stream>>>(xh, wvh, bv, proj, 8192, 1024, 1024);
  vpack_t<<<vpackBlocks, 256, 0, stream>>>(proj, vhb);

  // Attention
  flash_attn_wmma<<<dim3(32, 64), 128, 0, stream>>>(qhb, khb, vhb, maskf, attnh);

  // Output projection -> fp32 d_out
  gemm_bias_wmma<<<gemmGrid, 256, 0, stream>>>(attnh, woh, bo, out, 8192, 1024, 1024);
}
